// torch_sigma_2010044694686
// MI455X (gfx1250) — matrix operand order repaired and checked by reading, not compiled, not run on hardware
//
#include <hip/hip_runtime.h>

typedef __attribute__((ext_vector_type(16))) _Float16 v16h;
typedef __attribute__((ext_vector_type(8)))  _Float16 v8h;
typedef __attribute__((ext_vector_type(8)))  float    v8f;
typedef unsigned int v4u __attribute__((ext_vector_type(4)));
typedef int          v8i __attribute__((ext_vector_type(8)));
typedef int          v4i __attribute__((ext_vector_type(4)));

#define F1 512
#define ROWS_TOTAL (32 * 1024)     // B*N flattened rows
#define MTILE 64                   // rows per workgroup
#define HSTRIDE 520                // padded LDS row stride (halves): 1040 B
#define BSTRIDE 520                // padded B-panel row stride (halves)
#define PANEL_ROWS 32              // K-rows per TDM panel (one WMMA K-block)
#define SMEM_BYTES ((MTILE * HSTRIDE + 2 * PANEL_ROWS * BSTRIDE) * 2)  // 133120 B

// ---------------------------------------------------------------------------
// Prep: ro1_w [512(out) x 512(in)] f32 -> Wt [512(in) x 512(out)] f16
// (B-matrix layout: row k holds 512 contiguous N halves)
// ---------------------------------------------------------------------------
__global__ __launch_bounds__(256) void transpose_w_f16(
    const float* __restrict__ w, _Float16* __restrict__ wt)
{
    int idx = blockIdx.x * 256 + threadIdx.x;
    int k = idx >> 9;
    int n = idx & 511;
    wt[(size_t)k * F1 + n] = (_Float16)w[(size_t)n * F1 + k];
}

// ---------------------------------------------------------------------------
// TDM: async-load one 32x512 f16 panel of Wt into LDS, padded rows.
// D# group0: count=1 | lds_addr | global_addr | type=2
// D# group1: data_size=2B, pad 4 DWORDs per 256 DWORDs (1024B row -> 1040B),
//            tensor 512x512, tile 512x32, dim0 stride 512.
// ---------------------------------------------------------------------------
__device__ __forceinline__ void tdm_load_panel(unsigned lds_addr,
                                               const _Float16* gptr)
{
    unsigned long long ga = (unsigned long long)(uintptr_t)gptr;
    v4u g0;
    g0.x = 1u;                                   // count=1 (valid descriptor)
    g0.y = lds_addr;                             // LDS byte address
    g0.z = (unsigned)ga;                         // global_addr[31:0]
    g0.w = (unsigned)(ga >> 32) | (2u << 30);    // global_addr[56:32] | type=2
    v8i g1;
    g1[0] = (int)0x07D10000u;   // data_size=1(2B)<<16 | pad_en<<20 | intv=7<<22 | amt=3<<25
    g1[1] = (int)(512u << 16);  // tensor_dim0[15:0] in [31:16]
    g1[2] = (int)(512u << 16);  // tensor_dim0 hi=0 | tensor_dim1[15:0]
    g1[3] = (int)(512u << 16);  // tensor_dim1 hi=0 | tile_dim0=512
    g1[4] = PANEL_ROWS;         // tile_dim1=32 | tile_dim2=0
    g1[5] = 512;                // tensor_dim0_stride[31:0] = 512 elements
    g1[6] = 0;
    g1[7] = 0;
    v4i z4 = {0, 0, 0, 0};              // groups 2/3 unused (2D tile)
    v8i z8 = {0, 0, 0, 0, 0, 0, 0, 0};  // trailing group (unused)
    __builtin_amdgcn_tensor_load_to_lds(g0, g1, z4, z4, z8, 0);
}

// ---------------------------------------------------------------------------
// Fused live path:  Hf = relu(H @ fe_w^T + fe_b)        (K=3, VALU)
//                   r  = relu(Hf @ ro1_w^T + ro1_b)     (WMMA f16, TDM-fed)
//                   out= r @ ro2_w^T + ro2_b            (N=5, VALU)
// Wave tiling: 2 row-tiles x 8 n-tiles (B fragment reused by 2 WMMAs).
// ---------------------------------------------------------------------------
__global__ __launch_bounds__(256) void fused_head(
    const float* __restrict__ H,        // [32768, 3]
    const float* __restrict__ few,      // [512, 3]
    const float* __restrict__ feb,      // [512]
    const _Float16* __restrict__ wt,    // [512(in), 512(out)] f16
    const float* __restrict__ ro1b,     // [512]
    const float* __restrict__ ro2w,     // [5, 512]
    const float* __restrict__ ro2b,     // [5]
    float* __restrict__ out)            // [32768, 5]
{
    extern __shared__ __align__(16) _Float16 smem[];
    _Float16* sHf = smem;                              // MTILE x HSTRIDE
    _Float16* sB  = smem + MTILE * HSTRIDE;            // 2 x 32 x BSTRIDE

    const int tid = threadIdx.x;
    const int lane = tid & 31;
    const int wid  = tid >> 5;
    const int rowbase = blockIdx.x * MTILE;

    const unsigned ldsBase = (unsigned)(uintptr_t)(void*)smem;
    const unsigned ldsB0 = ldsBase + (unsigned)(MTILE * HSTRIDE * 2);
    const unsigned ldsB1 = ldsB0 + (unsigned)(PANEL_ROWS * BSTRIDE * 2);

    // Kick off the first B panel immediately; it streams while fe runs.
    if (wid == 0) tdm_load_panel(ldsB0, wt);

    // ---- Stage 1: feature extractor (K=3) -> Hf in LDS as f16 -------------
    for (int i = tid; i < MTILE * F1; i += 256) {
        const int m = i >> 9;
        const int c = i & 511;
        const float* hr = H + (size_t)(rowbase + m) * 3;
        float v = hr[0] * few[c * 3 + 0] + hr[1] * few[c * 3 + 1] +
                  hr[2] * few[c * 3 + 2] + feb[c];
        sHf[m * HSTRIDE + c] = (_Float16)(v > 0.f ? v : 0.f);
    }

    // ---- Stage 2: 64x512 @ 512x512 GEMM, WMMA fed from TDM-staged LDS -----
    const int mtbase = (wid & 1) * 32;        // two 16-row tiles: +0, +16
    const int nbase0 = (wid >> 1) * 128;      // 4 col-groups (8 n-tiles each)
    const int am   = lane & 15;               // A: matrix row
    const int akhi = (lane >> 4) << 3;        // A: K half-offset (0 or 8)

    v8f acc[2][8] = {};

    #pragma unroll 1
    for (int kb = 0; kb < 16; ++kb) {
        // Prefetch next panel into the other buffer; ensure current is done.
        if (kb < 15) {
            if (wid == 0)
                tdm_load_panel((kb & 1) ? ldsB0 : ldsB1,
                               wt + (size_t)(kb + 1) * PANEL_ROWS * F1);
            __builtin_amdgcn_s_wait_tensorcnt(1);
        } else {
            __builtin_amdgcn_s_wait_tensorcnt(0);
        }
        __syncthreads();   // panel kb visible to all waves (also covers fe)

        const int k0 = kb * 32;
        union Au { v16h v; v8h h[2]; };
        Au A0, A1;
        const _Float16* pA0 = &sHf[(mtbase + am) * HSTRIDE + k0 + akhi];
        const _Float16* pA1 = pA0 + 16 * HSTRIDE;
        A0.h[0] = *(const v8h*)(pA0);
        A0.h[1] = *(const v8h*)(pA0 + 16);
        A1.h[0] = *(const v8h*)(pA1);
        A1.h[1] = *(const v8h*)(pA1 + 16);

        const _Float16* pB = sB + (kb & 1) * PANEL_ROWS * BSTRIDE
                                + akhi * BSTRIDE + nbase0 + (lane & 15);
        #pragma unroll
        for (int t = 0; t < 8; ++t) {
            union { v16h v; v8h h[2]; } Bf;
            #pragma unroll
            for (int i = 0; i < 8; ++i) {
                Bf.v[i]     = pB[t * 16 + i * BSTRIDE];
                Bf.v[8 + i] = pB[t * 16 + (16 + i) * BSTRIDE];
            }
            acc[0][t] = __builtin_amdgcn_wmma_f32_16x16x32_f16(
                false, A0.v, false, Bf.v, (short)0, acc[0][t], false, false);
            acc[1][t] = __builtin_amdgcn_wmma_f32_16x16x32_f16(
                false, A1.v, false, Bf.v, (short)0, acc[1][t], false, false);
        }

        // Two-tile-deep pipelined schedule: 8 DS reads first (A0,A1 = 4 b128; one B = 16 u16),
        // then {WMMA pair t, 2 DS reads} x6, then the last 2 pairs.
        // With B(t+1) already in flight when pair t waits, the dscnt wait
        // relaxes from 0 to <=2 and loads get ~2 WMMA pairs of cover.
        __builtin_amdgcn_sched_group_barrier(0x100, 8, 0);   // DS read x8
        #pragma unroll
        for (int s = 0; s < 6; ++s) {
            __builtin_amdgcn_sched_group_barrier(0x008, 2, 0); // WMMA x2
            __builtin_amdgcn_sched_group_barrier(0x100, 2, 0); // DS read x2
        }
        __builtin_amdgcn_sched_group_barrier(0x008, 4, 0);     // WMMA x4

        __syncthreads();   // all waves done with panel kb before overwrite
    }

    // ---- Epilogue: bias + relu, write r back over sHf as f16 --------------
    const int dn   = lane & 15;
    const int dmhi = (lane >> 4) << 3;
    #pragma unroll
    for (int rt = 0; rt < 2; ++rt) {
        #pragma unroll
        for (int t = 0; t < 8; ++t) {
            const int n = nbase0 + t * 16 + dn;
            const float bias = ro1b[n];
            #pragma unroll
            for (int v = 0; v < 8; ++v) {
                float r = acc[rt][t][v] + bias;
                sHf[(mtbase + rt * 16 + dmhi + v) * HSTRIDE + n] =
                    (_Float16)(r > 0.f ? r : 0.f);
            }
        }
    }
    __syncthreads();

    // ---- Stage 3: ro2 head, 64 rows x 5 outputs, K=512 --------------------
    if (tid < MTILE * 5) {
        const int m = tid / 5;
        const int j = tid - m * 5;
        const _Float16* rr = &sHf[m * HSTRIDE];
        const float* wj = ro2w + j * F1;
        float s = ro2b[j];
        #pragma unroll 8
        for (int k = 0; k < F1; ++k)
            s += (float)rr[k] * wj[k];
        out[(size_t)(rowbase + m) * 5 + j] = s;
    }
}

// ---------------------------------------------------------------------------
extern "C" void kernel_launch(void* const* d_in, const int* in_sizes, int n_in,
                              void* d_out, int out_size, void* d_ws, size_t ws_size,
                              hipStream_t stream)
{
    // 0:H 1:fe_w 2:fe_b ... 14:ro1_w 15:ro1_b 16:ro2_w 17:ro2_b
    const float* H    = (const float*)d_in[0];
    const float* few  = (const float*)d_in[1];
    const float* feb  = (const float*)d_in[2];
    const float* ro1w = (const float*)d_in[14];
    const float* ro1b = (const float*)d_in[15];
    const float* ro2w = (const float*)d_in[16];
    const float* ro2b = (const float*)d_in[17];
    float* out = (float*)d_out;

    _Float16* wt = (_Float16*)d_ws;   // 512 KB scratch

    transpose_w_f16<<<(F1 * F1) / 256, 256, 0, stream>>>(ro1w, wt);
    fused_head<<<ROWS_TOTAL / MTILE, 256, SMEM_BYTES, stream>>>(
        H, few, feb, wt, ro1b, ro2w, ro2b, out);
}
